// _HeteroNodeLayer_90941637526224
// MI455X (gfx1250) — matrix operand order repaired and checked by reading, not compiled, not run on hardware
//
#include <hip/hip_runtime.h>

typedef __bf16 bf16;
typedef __attribute__((ext_vector_type(16))) __bf16 bf16x16;
typedef __attribute__((ext_vector_type(8)))  __bf16 bf16x8;
typedef __attribute__((ext_vector_type(8)))  float  f32x8;
typedef __attribute__((ext_vector_type(4)))  float  f32x4;

#define NHEADS 8
#define DH     16
#define DMODEL 128
#define NTOK   2048
#define BATCH  4
#define DFF    512

enum { MODE_QK = 0, MODE_VT = 1, MODE_ADDF32 = 2, MODE_RELUBF16 = 3 };

__device__ __forceinline__ float nan2num(float v) {
    if (v != v) return 0.0f;
    const float FMAX = 3.4028234663852886e38f;
    if (v >  FMAX) return  FMAX;
    if (v < -FMAX) return -FMAX;
    return v;
}

// ---------------------------------------------------------------------------
// h (f32) -> residual copy (f32) + bf16 activation copy
// ---------------------------------------------------------------------------
__global__ void cvt_h_kernel(const float* __restrict__ h, float* __restrict__ res,
                             bf16* __restrict__ hb, int n) {
    int i = blockIdx.x * blockDim.x + threadIdx.x;
    if (i < n) {
        float v = h[i];
        res[i] = v;
        hb[i]  = (bf16)v;
    }
}

// W [K,N] f32 row-major  ->  Wt [N,K] bf16 row-major (transposed for B-frags)
__global__ void wt_kernel(const float* __restrict__ W, bf16* __restrict__ Wt,
                          int K, int N) {
    int i = blockIdx.x * blockDim.x + threadIdx.x;
    if (i < K * N) {
        int k = i / N, n = i % N;
        Wt[(size_t)n * K + k] = (bf16)W[i];
    }
}

// ---------------------------------------------------------------------------
// bf16 WMMA GEMM: one wave computes a 16x64 output strip (4 accumulators,
// A-fragment reused across the 4 N-subtiles). K-loop step 32 via
// v_wmma_f32_16x16x32_bf16 (f32 accumulate).
// ---------------------------------------------------------------------------
__global__ void __launch_bounds__(128)
gemm_wmma(const bf16* __restrict__ A, const bf16* __restrict__ Wt,
          const float* __restrict__ bias,
          float* __restrict__ outF, bf16* __restrict__ outH,
          int M, int N, int K, int mode,
          const int* __restrict__ gatherIdx, int Mb, int tokPerBatch,
          const int* __restrict__ scatterIdx, int ldOut) {
    const int lane = threadIdx.x & 31;
    const int wave = threadIdx.x >> 5;
    const int stripsN = N >> 6;                 // 64-wide strips
    const int strips  = (M >> 4) * stripsN;
    const int strip   = blockIdx.x * 4 + wave;
    if (strip >= strips) return;                // wave-uniform; EXEC stays all-1s

    const int tm  = strip / stripsN, tsn = strip % stripsN;
    const int g   = lane >> 4, li = lane & 15;
    const int khalf = g * 8;

    // A row (optionally gathered per-batch: h[:, idx])
    int mrow = tm * 16 + li;
    long arow;
    if (gatherIdx) {
        int bb = mrow / Mb, ii = mrow % Mb;
        arow = (long)bb * tokPerBatch + gatherIdx[ii];
    } else {
        arow = mrow;
    }
    const bf16* Arow  = A  + (size_t)arow * K;
    const bf16* Wrow0 = Wt + (size_t)(tsn * 64 + li) * K;   // sub-tile j adds 16*K

    f32x8 acc0 = {}, acc1 = {}, acc2 = {}, acc3 = {};
    for (int kb = 0; kb < K; kb += 32) {
        union { bf16x16 v; bf16x8 h2[2]; } a, b;
        // A frag (16x32): lanes 0-15 -> K 0..15, lanes 16-31 -> 16..31
        a.h2[0] = *(const bf16x8*)(Arow + kb + 2 * khalf);
        a.h2[1] = *(const bf16x8*)(Arow + kb + 2 * khalf + 8);
        // B frags (32x16): lane holds one column, K = kb + 16*g .. +15 contiguous
        b.v  = *(const bf16x16*)(Wrow0 + 0 * 16 * K + kb + g * 16);
        acc0 = __builtin_amdgcn_wmma_f32_16x16x32_bf16(false, a.v, false, b.v,
                                                       (short)0, acc0, false, false);
        b.v  = *(const bf16x16*)(Wrow0 + 1 * 16 * K + kb + g * 16);
        acc1 = __builtin_amdgcn_wmma_f32_16x16x32_bf16(false, a.v, false, b.v,
                                                       (short)0, acc1, false, false);
        b.v  = *(const bf16x16*)(Wrow0 + 2 * 16 * K + kb + g * 16);
        acc2 = __builtin_amdgcn_wmma_f32_16x16x32_bf16(false, a.v, false, b.v,
                                                       (short)0, acc2, false, false);
        b.v  = *(const bf16x16*)(Wrow0 + 3 * 16 * K + kb + g * 16);
        acc3 = __builtin_amdgcn_wmma_f32_16x16x32_bf16(false, a.v, false, b.v,
                                                       (short)0, acc3, false, false);
    }

    // C/D layout: VGPR r -> row tm*16 + r + 8*g, col = ncol_j
    f32x8 accs[4] = {acc0, acc1, acc2, acc3};
#pragma unroll
    for (int j = 0; j < 4; ++j) {
        const int ncol = tsn * 64 + j * 16 + li;
        const float bv = bias[ncol];
        if (mode == MODE_QK || mode == MODE_VT) {
            const int hh = ncol >> 4, d = ncol & 15;
#pragma unroll
            for (int r = 0; r < 8; ++r) {
                int m  = tm * 16 + r + 8 * g;
                int bb = m / Mb, ii = m % Mb;
                float v = accs[j][r] + bv;
                size_t idx;
                if (mode == MODE_QK)
                    idx = (((size_t)bb * NHEADS + hh) * Mb + ii) * DH + d;        // [B,H,n,dh]
                else
                    idx = (((size_t)bb * NHEADS + hh) * DH + d) * (size_t)Mb + ii; // [B,H,dh,n]
                outH[idx] = (bf16)v;
            }
        } else if (mode == MODE_ADDF32) {
#pragma unroll
            for (int r = 0; r < 8; ++r) {
                int m = tm * 16 + r + 8 * g;
                long row;
                if (scatterIdx) {
                    int bb = m / Mb, ii = m % Mb;
                    row = (long)bb * tokPerBatch + scatterIdx[ii];
                } else {
                    row = m;
                }
                outF[(size_t)row * ldOut + ncol] += accs[j][r] + bv;
            }
        } else { // MODE_RELUBF16
#pragma unroll
            for (int r = 0; r < 8; ++r) {
                int m = tm * 16 + r + 8 * g;
                outH[(size_t)m * N + ncol] = (bf16)fmaxf(accs[j][r] + bv, 0.0f);
            }
        }
    }
}

// ---------------------------------------------------------------------------
// Flash attention, dh = 16. One wave handles 16 queries of one (b, head),
// marching the key dimension in blocks of 64:
//   4 score WMMAs (dh zero-padded to 32) -> one online-softmax update
//   (shfl_xor row reductions within 16-lane groups) -> P staged through LDS
//   (C-layout -> A-layout relayout) -> 2 P@V WMMAs.
// Q: [B,H,nq,16], K: [B,H,nk,16], Vt: [B,H,16,nk], all bf16.
// ---------------------------------------------------------------------------
__global__ void __launch_bounds__(128)
flash_wmma(const bf16* __restrict__ Q, const bf16* __restrict__ Kmat,
           const bf16* __restrict__ Vt, bf16* __restrict__ out,
           int nq, int nk) {
    __shared__ bf16 ldsP[4][4][16][16];   // [wave][key-quarter][row m][col]

    const int lane = threadIdx.x & 31;
    const int wave = threadIdx.x >> 5;
    const int g = lane >> 4, li = lane & 15;
    const int qtile = blockIdx.x * 4 + wave;
    if (qtile * 16 >= nq) return;
    const int hd = blockIdx.y;
    const int b  = blockIdx.z;

    const size_t bh = (size_t)b * NHEADS + hd;
    const bf16* Qh = Q    + bh * (size_t)nq * DH;
    const bf16* Kh = Kmat + bh * (size_t)nk * DH;
    const bf16* Vh = Vt   + bh * (size_t)DH * nk;

    // Q A-frag (16 x 32, K>=16 padded with zeros)
    union { bf16x16 v; bf16x8 h2[2]; } aQ = {};
    {
        int q = qtile * 16 + li;
        if (g == 0) aQ.v = *(const bf16x16*)(Qh + (size_t)q * DH);
        // g==1 lanes stay zero (padding of dh to 32)
    }

    f32x8 acc = {};
    float rmax[8], rsum[8];
#pragma unroll
    for (int r = 0; r < 8; ++r) { rmax[r] = -3.0e38f; rsum[r] = 0.0f; }

    union { bf16x16 v; bf16x8 h2[2]; } bK = {};  // g==1 lanes: padded dh rows, stay 0
    const f32x8 zc = {};

    for (int kb = 0; kb < nk; kb += 64) {
        if (kb + 64 < nk) {  // prefetch next key/value block (global_prefetch_b8)
            __builtin_prefetch(Kh + (size_t)(kb + 64 + li) * DH, 0, 3);
            __builtin_prefetch(Vh + (size_t)li * nk + kb + 64, 0, 3);
        }

        // 4 score tiles: keys kb+16t .. kb+16t+15
        f32x8 s[4];
#pragma unroll
        for (int t = 0; t < 4; ++t) {
            if (g == 0)
                bK.v = *(const bf16x16*)(Kh + (size_t)(kb + t * 16 + li) * DH);
            s[t] = __builtin_amdgcn_wmma_f32_16x16x32_bf16(false, aQ.v, false, bK.v,
                                                           (short)0, zc, false, false);
        }

        // online softmax over the 64-key block
#pragma unroll
        for (int r = 0; r < 8; ++r) {
            float a0 = s[0][r] * 0.25f;           // 1/sqrt(dh)
            float a1 = s[1][r] * 0.25f;
            float a2 = s[2][r] * 0.25f;
            float a3 = s[3][r] * 0.25f;
            float t = fmaxf(fmaxf(a0, a1), fmaxf(a2, a3));
            t = fmaxf(t, __shfl_xor(t, 1));
            t = fmaxf(t, __shfl_xor(t, 2));
            t = fmaxf(t, __shfl_xor(t, 4));
            t = fmaxf(t, __shfl_xor(t, 8));
            float nm   = fmaxf(rmax[r], t);
            float corr = __expf(rmax[r] - nm);
            float p0 = __expf(a0 - nm);
            float p1 = __expf(a1 - nm);
            float p2 = __expf(a2 - nm);
            float p3 = __expf(a3 - nm);
            float ps = (p0 + p1) + (p2 + p3);
            ps += __shfl_xor(ps, 1);
            ps += __shfl_xor(ps, 2);
            ps += __shfl_xor(ps, 4);
            ps += __shfl_xor(ps, 8);
            rsum[r] = rsum[r] * corr + ps;
            rmax[r] = nm;
            acc[r]  = acc[r] * corr;
            // stage P (C-layout) into LDS, row-major per 16-key quarter
            const int m = r + 8 * g;
            ldsP[wave][0][m][li] = (bf16)p0;
            ldsP[wave][1][m][li] = (bf16)p1;
            ldsP[wave][2][m][li] = (bf16)p2;
            ldsP[wave][3][m][li] = (bf16)p3;
        }

        // P A-frags (16 x 32) from LDS, V B-frags (32 x 16) from transposed V
        union { bf16x16 v; bf16x8 h2[2]; } aP, bV;
        aP.h2[0] = *(const bf16x8*)&ldsP[wave][g][li][0];
        aP.h2[1] = *(const bf16x8*)&ldsP[wave][g][li][8];
        bV.v     = *(const bf16x16*)(Vh + (size_t)li * nk + kb + 16 * g);
        acc = __builtin_amdgcn_wmma_f32_16x16x32_bf16(false, aP.v, false, bV.v,
                                                      (short)0, acc, false, false);
        aP.h2[0] = *(const bf16x8*)&ldsP[wave][2 + g][li][0];
        aP.h2[1] = *(const bf16x8*)&ldsP[wave][2 + g][li][8];
        bV.v     = *(const bf16x16*)(Vh + (size_t)li * nk + kb + 32 + 16 * g);
        acc = __builtin_amdgcn_wmma_f32_16x16x32_bf16(false, aP.v, false, bV.v,
                                                      (short)0, acc, false, false);
    }

#pragma unroll
    for (int r = 0; r < 8; ++r) {
        float v = acc[r] / rsum[r];
        int q = qtile * 16 + r + 8 * g;
        out[((size_t)b * nq + q) * DMODEL + hd * DH + li] = (bf16)v;
    }
}

// ---------------------------------------------------------------------------
// LayerNorm (biased variance, eps=1e-5) + nan_to_num.
// One wave per row of 128; optionally also emits a bf16 copy.
// ---------------------------------------------------------------------------
__global__ void __launch_bounds__(256)
ln_kernel(const float* __restrict__ x, const float* __restrict__ gam,
          const float* __restrict__ bet, float* __restrict__ yF,
          bf16* __restrict__ yH, int T) {
    const int lane = threadIdx.x & 31;
    const int wave = threadIdx.x >> 5;
    const int row  = blockIdx.x * 8 + wave;
    if (row >= T) return;

    const float* xr = x + (size_t)row * DMODEL;
    f32x4 v = *(const f32x4*)(xr + lane * 4);
    float s = v[0] + v[1] + v[2] + v[3];
    float q = v[0] * v[0] + v[1] * v[1] + v[2] * v[2] + v[3] * v[3];
#pragma unroll
    for (int m = 1; m < 32; m <<= 1) {
        s += __shfl_xor(s, m);
        q += __shfl_xor(q, m);
    }
    const float mean = s * (1.0f / DMODEL);
    const float var  = q * (1.0f / DMODEL) - mean * mean;
    const float rstd = rsqrtf(var + 1e-5f);
#pragma unroll
    for (int j = 0; j < 4; ++j) {
        int c = lane * 4 + j;
        float y = (v[j] - mean) * rstd * gam[c] + bet[c];
        y = nan2num(y);
        yF[(size_t)row * DMODEL + c] = y;
        if (yH) yH[(size_t)row * DMODEL + c] = (bf16)y;
    }
}

// ---------------------------------------------------------------------------
// Host orchestration
// ---------------------------------------------------------------------------
extern "C" void kernel_launch(void* const* d_in, const int* in_sizes, int n_in,
                              void* d_out, int out_size, void* d_ws, size_t ws_size,
                              hipStream_t stream) {
    const float* h     = (const float*)d_in[0];
    const int*   l_idx = (const int*)  d_in[1];
    const int*   b_idx = (const int*)  d_in[2];

    const float* Wm[12]; const float* Bv[12];
    for (int p = 0; p < 3; ++p)
        for (int j = 0; j < 4; ++j) {
            Wm[p * 4 + j] = (const float*)d_in[3 + p * 8 + j * 2];
            Bv[p * 4 + j] = (const float*)d_in[3 + p * 8 + j * 2 + 1];
        }
    const float* ff_w1 = (const float*)d_in[27];
    const float* ff_b1 = (const float*)d_in[28];
    const float* ff_w2 = (const float*)d_in[29];
    const float* ff_b2 = (const float*)d_in[30];
    const float* n1_g  = (const float*)d_in[31];
    const float* n1_b  = (const float*)d_in[32];
    const float* n2_g  = (const float*)d_in[33];
    const float* n2_b  = (const float*)d_in[34];

    const int T  = BATCH * NTOK;          // 8192 tokens
    const int Th = BATCH * (NTOK / 2);    // 4096 tokens per split

    // --- workspace carve-up (256B aligned) ---
    char* base = (char*)d_ws;
    size_t off = 0;
    auto alloc = [&](size_t bytes) -> void* {
        void* p = base + off;
        off = (off + bytes + 255) & ~(size_t)255;
        return p;
    };
    bf16*  hb      = (bf16*) alloc((size_t)T * DMODEL * 2);
    float* res     = (float*)alloc((size_t)T * DMODEL * 4);
    float* res2    = (float*)alloc((size_t)T * DMODEL * 4);
    bf16*  xb      = (bf16*) alloc((size_t)T * DMODEL * 2);
    bf16*  f1      = (bf16*) alloc((size_t)T * DFF * 2);
    bf16*  Wt12    = (bf16*) alloc((size_t)12 * DMODEL * DMODEL * 2);
    bf16*  W1t     = (bf16*) alloc((size_t)DFF * DMODEL * 2);
    bf16*  W2t     = (bf16*) alloc((size_t)DMODEL * DFF * 2);
    bf16*  Qsa     = (bf16*) alloc((size_t)T * DMODEL * 2);
    bf16*  Ksa     = (bf16*) alloc((size_t)T * DMODEL * 2);
    bf16*  Vsa     = (bf16*) alloc((size_t)T * DMODEL * 2);
    bf16*  Ql      = (bf16*) alloc((size_t)Th * DMODEL * 2);
    bf16*  Kb      = (bf16*) alloc((size_t)Th * DMODEL * 2);
    bf16*  Vb      = (bf16*) alloc((size_t)Th * DMODEL * 2);
    bf16*  Qb      = (bf16*) alloc((size_t)Th * DMODEL * 2);
    bf16*  Kl      = (bf16*) alloc((size_t)Th * DMODEL * 2);
    bf16*  Vl      = (bf16*) alloc((size_t)Th * DMODEL * 2);
    bf16*  attn_sa = (bf16*) alloc((size_t)T * DMODEL * 2);
    bf16*  attn_l  = (bf16*) alloc((size_t)Th * DMODEL * 2);
    bf16*  attn_b  = (bf16*) alloc((size_t)Th * DMODEL * 2);

    // --- 1. convert activations + residual, transpose weights ---
    {
        int n = T * DMODEL;
        cvt_h_kernel<<<(n + 255) / 256, 256, 0, stream>>>(h, res, hb, n);
    }
    for (int i = 0; i < 12; ++i)
        wt_kernel<<<(DMODEL * DMODEL + 255) / 256, 256, 0, stream>>>(
            Wm[i], Wt12 + (size_t)i * DMODEL * DMODEL, DMODEL, DMODEL);
    wt_kernel<<<(DMODEL * DFF + 255) / 256, 256, 0, stream>>>(ff_w1, W1t, DMODEL, DFF);
    wt_kernel<<<(DFF * DMODEL + 255) / 256, 256, 0, stream>>>(ff_w2, W2t, DFF, DMODEL);

    auto gemm = [&](const bf16* A, const bf16* Wt, const float* bias,
                    float* oF, bf16* oH, int M, int N, int K, int mode,
                    const int* gi, int Mb, const int* si, int ldo) {
        int strips = (M / 16) * (N / 64);
        gemm_wmma<<<(strips + 3) / 4, 128, 0, stream>>>(
            A, Wt, bias, oF, oH, M, N, K, mode, gi, Mb, NTOK, si, ldo);
    };

    const size_t WSZ = (size_t)DMODEL * DMODEL;
    // --- 2. Q/K/V projections (head-major; V transposed) ---
    gemm(hb, Wt12 +  0 * WSZ, Bv[0],  nullptr, Qsa, T,  DMODEL, DMODEL, MODE_QK, nullptr, NTOK,     nullptr, 0);
    gemm(hb, Wt12 +  1 * WSZ, Bv[1],  nullptr, Ksa, T,  DMODEL, DMODEL, MODE_QK, nullptr, NTOK,     nullptr, 0);
    gemm(hb, Wt12 +  2 * WSZ, Bv[2],  nullptr, Vsa, T,  DMODEL, DMODEL, MODE_VT, nullptr, NTOK,     nullptr, 0);
    gemm(hb, Wt12 +  4 * WSZ, Bv[4],  nullptr, Ql,  Th, DMODEL, DMODEL, MODE_QK, l_idx,   NTOK / 2, nullptr, 0);
    gemm(hb, Wt12 +  5 * WSZ, Bv[5],  nullptr, Kb,  Th, DMODEL, DMODEL, MODE_QK, b_idx,   NTOK / 2, nullptr, 0);
    gemm(hb, Wt12 +  6 * WSZ, Bv[6],  nullptr, Vb,  Th, DMODEL, DMODEL, MODE_VT, b_idx,   NTOK / 2, nullptr, 0);
    gemm(hb, Wt12 +  8 * WSZ, Bv[8],  nullptr, Qb,  Th, DMODEL, DMODEL, MODE_QK, b_idx,   NTOK / 2, nullptr, 0);
    gemm(hb, Wt12 +  9 * WSZ, Bv[9],  nullptr, Kl,  Th, DMODEL, DMODEL, MODE_QK, l_idx,   NTOK / 2, nullptr, 0);
    gemm(hb, Wt12 + 10 * WSZ, Bv[10], nullptr, Vl,  Th, DMODEL, DMODEL, MODE_VT, l_idx,   NTOK / 2, nullptr, 0);

    // --- 3. flash attention (SA, l->b, b->l) ---
    flash_wmma<<<dim3(NTOK / 16 / 4, NHEADS, BATCH), 128, 0, stream>>>(Qsa, Ksa, Vsa, attn_sa, NTOK, NTOK);
    flash_wmma<<<dim3((NTOK / 2) / 16 / 4, NHEADS, BATCH), 128, 0, stream>>>(Ql, Kb, Vb, attn_l, NTOK / 2, NTOK / 2);
    flash_wmma<<<dim3((NTOK / 2) / 16 / 4, NHEADS, BATCH), 128, 0, stream>>>(Qb, Kl, Vl, attn_b, NTOK / 2, NTOK / 2);

    // --- 4. O-projections accumulate into residual (scattered for splits) ---
    gemm(attn_sa, Wt12 +  3 * WSZ, Bv[3],  res, nullptr, T,  DMODEL, DMODEL, MODE_ADDF32, nullptr, NTOK,     nullptr, DMODEL);
    gemm(attn_l,  Wt12 +  7 * WSZ, Bv[7],  res, nullptr, Th, DMODEL, DMODEL, MODE_ADDF32, nullptr, NTOK / 2, l_idx,   DMODEL);
    gemm(attn_b,  Wt12 + 11 * WSZ, Bv[11], res, nullptr, Th, DMODEL, DMODEL, MODE_ADDF32, nullptr, NTOK / 2, b_idx,   DMODEL);

    // --- 5. LN1 (-> fp32 residual2 + bf16 FFN input) ---
    ln_kernel<<<T / 8, 256, 0, stream>>>(res, n1_g, n1_b, res2, xb, T);

    // --- 6. FFN ---
    gemm(xb, W1t, ff_b1, nullptr, f1,   T, DFF,    DMODEL, MODE_RELUBF16, nullptr, T, nullptr, 0);
    gemm(f1, W2t, ff_b2, res2,   nullptr, T, DMODEL, DFF,  MODE_ADDF32,   nullptr, T, nullptr, DMODEL);

    // --- 7. LN2 -> output ---
    ln_kernel<<<T / 8, 256, 0, stream>>>(res2, n2_g, n2_b, (float*)d_out, nullptr, T);

    (void)in_sizes; (void)n_in; (void)out_size; (void)ws_size;
}
